// GRUEncDecWithHead_79370995631069
// MI455X (gfx1250) — matrix operand order repaired and checked by reading, not compiled, not run on hardware
//
#include <hip/hip_runtime.h>
#include <hip/hip_bf16.h>
#include <math.h>

typedef __attribute__((ext_vector_type(16))) _Float16 v16h;
typedef __attribute__((ext_vector_type(8)))  float    v8f;

#define GRU_H   64
#define NGATE   192   // 3*H
#define MB      16    // batch rows per workgroup (one WMMA M tile)
#define ENC_T   30
#define DEC_T   20

// conv2 implicit GEMM geometry: M=13*13=169 (pad 176), K=4*3*3=36 (pad 64), N=10 (pad 16)
#define C2_M      169
#define C2_MPAD   176
#define C2_K      36
#define C2_KPAD   64
#define C2_N      10

// ---------------------------------------------------------------------------
// WMMA operand loaders (layouts per CDNA5 ISA 7.12.2, wave32)
// ---------------------------------------------------------------------------

// A-matrix 16x32 f16 from LDS. Row-major LDS tile, row stride ldK halves.
// Lane l holds row M = l&15; element j holds K = j + 8*(j>=8) + 8*(l>=16).
__device__ __forceinline__ v16h load_A16x32(const _Float16* lds, int ldK, int lane) {
    int row   = lane & 15;
    int kbase = (lane >> 4) << 3;          // 0 or 8
    const _Float16* p = lds + row * ldK;
    v16h a;
#pragma unroll
    for (int j = 0; j < 8; ++j) a[j] = p[kbase + j];
#pragma unroll
    for (int j = 0; j < 8; ++j) a[8 + j] = p[16 + kbase + j];
    return a;
}

// B-matrix 32x16 f16 built from a row-major f32 weight W[N][ldk]:
// B[k][n] = W[n][k].  Lane l holds column N = ncol + (l&15);
// element h holds K = koff + h + 8*(h>=8) + 8*(l>=16).
// k >= Kvalid or n >= Nvalid -> 0 (padding; also guards OOB reads of W).
__device__ __forceinline__ v16h load_B32x16(const float* W, int ldk, int ncol,
                                            int koff, int Kvalid, int Nvalid,
                                            int lane) {
    int n  = ncol + (lane & 15);
    int kb = koff + ((lane >> 4) << 3);    // +8 on upper half-wave
    v16h b;
#pragma unroll
    for (int h = 0; h < 16; ++h) {
        int k = kb + h + ((h >> 3) << 3);
        b[h] = (k < Kvalid && n < Nvalid) ? (_Float16)W[n * ldk + k] : (_Float16)0.f;
    }
    return b;
}

// C/D f32 16x16 accumulator -> LDS [16][ldn]: vreg r holds row r + 8*(l>=16),
// column lane&15.
__device__ __forceinline__ void store_acc16x16(float* lds, int ldn, int ncol,
                                               v8f acc, int lane) {
    int n     = ncol + (lane & 15);
    int rbase = (lane >> 4) << 3;
#pragma unroll
    for (int r = 0; r < 8; ++r) lds[(rbase + r) * ldn + n] = acc[r];
}

__device__ __forceinline__ float sigmoidf_(float x) { return 1.f / (1.f + expf(-x)); }

// Pointwise GRU gate math over the MB x H state (PyTorch gate order r,z,n).
__device__ __forceinline__ void gru_update(const float* __restrict__ gi,
                                           const float* __restrict__ gh,
                                           float* __restrict__ h32,
                                           _Float16* __restrict__ h16,
                                           const float* __restrict__ bih,
                                           const float* __restrict__ bhh,
                                           int tid) {
    for (int i = tid; i < MB * GRU_H; i += 128) {
        int m = i >> 6, c = i & 63;
        const float* gim = gi + m * NGATE;
        const float* ghm = gh + m * NGATE;
        float r = sigmoidf_(gim[c]       + bih[c]       + ghm[c]       + bhh[c]);
        float z = sigmoidf_(gim[64 + c]  + bih[64 + c]  + ghm[64 + c]  + bhh[64 + c]);
        float n = tanhf    (gim[128 + c] + bih[128 + c] + r * (ghm[128 + c] + bhh[128 + c]));
        float hn = (1.f - z) * n + z * h32[i];
        h32[i] = hn;
        h16[i] = (_Float16)hn;
    }
}

// One conv2 WMMA M-tile: A panel rows [mt*16, mt*16+16) of im2col, K=64 in two
// k-steps, f32 accumulate, result -> c2 panel [176][16].
__device__ __forceinline__ void conv2_wmma_tile(const _Float16* __restrict__ im2col,
                                                float* __restrict__ c2,
                                                v16h b0, v16h b1, int mt, int lane) {
    const _Float16* base = im2col + mt * 16 * C2_KPAD;
    v16h a0 = load_A16x32(base,      C2_KPAD, lane);
    v16h a1 = load_A16x32(base + 32, C2_KPAD, lane);
    v8f acc = {};
    acc = __builtin_amdgcn_wmma_f32_16x16x32_f16(false, a0, false, b0,
                                                 (short)0, acc, false, false);
    acc = __builtin_amdgcn_wmma_f32_16x16x32_f16(false, a1, false, b1,
                                                 (short)0, acc, false, false);
    store_acc16x16(c2 + mt * 16 * 16, 16, 0, acc, lane);
}

// ---------------------------------------------------------------------------
// Kernel 1: ConvEncoder head -> 5-dim embedding, one workgroup per crop.
// conv(1->4,3x3)+relu+pool2 (VALU; K=9/N=4 too small for WMMA tiles)
// conv(4->10,3x3)+relu+pool2 (implicit-GEMM via v_wmma_f32_16x16x32_f16)
// fc(360->5) (VALU reduction)
// ---------------------------------------------------------------------------
__global__ __launch_bounds__(256) void conv_head_kernel(
    const float* __restrict__ head,     // [B,31,32,32]
    const float* __restrict__ w1, const float* __restrict__ b1,
    const float* __restrict__ w2, const float* __restrict__ b2,
    const float* __restrict__ fcw, const float* __restrict__ fcb,
    float* __restrict__ emb)            // [B*30, 5]
{
    __shared__ float    s_img[32 * 32];
    __shared__ float    s_w1[36], s_b1[4], s_b2[10];
    __shared__ float    s_fw[1800], s_fb[5];
    __shared__ _Float16 s_p1h[4 * 15 * 15];          // conv1+pool, [c][15][15], f16
    __shared__ _Float16 s_col[C2_MPAD * C2_KPAD];    // im2col A panel (zero padded)
    __shared__ float    s_c2[C2_MPAD * 16];          // conv2 pre-activation panel
    __shared__ float    s_p2[360];                   // conv2+pool, [c][6][6]
    __shared__ float    s_red[5 * 32];

    const int tid  = threadIdx.x;
    const int lane = tid & 31;
    const int wave = tid >> 5;
    const int img = blockIdx.x;          // b*30 + t
    const int b = img / 30, t = img % 30;

    const float* ip = head + ((size_t)b * 31 + t) * 1024;
    for (int i = tid; i < 1024; i += 256) s_img[i] = ip[i];
    for (int i = tid; i < 36;   i += 256) s_w1[i] = w1[i];
    for (int i = tid; i < 4;    i += 256) s_b1[i] = b1[i];
    for (int i = tid; i < 10;   i += 256) s_b2[i] = b2[i];
    for (int i = tid; i < 1800; i += 256) s_fw[i] = fcw[i];
    for (int i = tid; i < 5;    i += 256) s_fb[i] = fcb[i];

    // conv2 B panel (w2^T, 36x10 zero-padded to 64x16) -> VGPRs, from global
    v16h bc0 = load_B32x16(w2, C2_K, 0, 0,  C2_K, C2_N, lane);
    v16h bc1 = load_B32x16(w2, C2_K, 0, 32, C2_K, C2_N, lane);
    __syncthreads();

    // conv1 + relu + 2x2 maxpool  (30x30 -> 15x15, 4 channels), f16 out
    for (int idx = tid; idx < 900; idx += 256) {
        int c = idx / 225, rem = idx % 225, py = rem / 15, px = rem % 15;
        const float* wc = s_w1 + c * 9;
        float mx = -1e30f;
#pragma unroll
        for (int dy = 0; dy < 2; ++dy) {
#pragma unroll
            for (int dx = 0; dx < 2; ++dx) {
                int y = 2 * py + dy, x = 2 * px + dx;
                float acc = s_b1[c];
#pragma unroll
                for (int i = 0; i < 3; ++i) {
#pragma unroll
                    for (int j = 0; j < 3; ++j)
                        acc += s_img[(y + i) * 32 + (x + j)] * wc[i * 3 + j];
                }
                mx = fmaxf(mx, fmaxf(acc, 0.f));
            }
        }
        s_p1h[idx] = (_Float16)mx;
    }
    __syncthreads();

    // im2col: A[m][k], m = y*13+x over 13x13 outputs, k = (ic*3+ky)*3+kx
    for (int idx = tid; idx < C2_MPAD * C2_KPAD; idx += 256) {
        int m = idx >> 6, k = idx & 63;
        _Float16 v = (_Float16)0.f;
        if (m < C2_M && k < C2_K) {
            int y = m / 13, x = m % 13;
            int ic = k / 9, r = k % 9, ky = r / 3, kx = r % 3;
            v = s_p1h[ic * 225 + (y + ky) * 15 + (x + kx)];
        }
        s_col[idx] = v;
    }
    __syncthreads();

    // conv2 GEMM: 11 M-tiles over 8 waves (waves 0..2 take a second tile)
    conv2_wmma_tile(s_col, s_c2, bc0, bc1, wave, lane);
    if (wave < 3) conv2_wmma_tile(s_col, s_c2, bc0, bc1, wave + 8, lane);
    __syncthreads();

    // bias + relu + 2x2 maxpool  (13x13 -> 6x6, 10 channels)
    for (int idx = tid; idx < 360; idx += 256) {
        int c = idx / 36, rem = idx % 36, py = rem / 6, px = rem % 6;
        float bias = s_b2[c];
        float mx = -1e30f;
#pragma unroll
        for (int dy = 0; dy < 2; ++dy) {
#pragma unroll
            for (int dx = 0; dx < 2; ++dx) {
                int y = 2 * py + dy, x = 2 * px + dx;
                float acc = s_c2[(y * 13 + x) * 16 + c] + bias;
                mx = fmaxf(mx, fmaxf(acc, 0.f));
            }
        }
        s_p2[idx] = mx;           // layout c*36 + py*6 + px (matches fc_w)
    }
    __syncthreads();

    // fc 360 -> 5 : 5 outputs x 32-lane partial sums, LDS reduce
    if (tid < 160) {
        int o = tid >> 5, l = tid & 31;
        float p = 0.f;
        for (int k = l; k < 360; k += 32) p += s_p2[k] * s_fw[o * 360 + k];
        s_red[tid] = p;
    }
    __syncthreads();
    if (tid < 5) {
        float a = s_fb[tid];
#pragma unroll
        for (int l = 0; l < 32; ++l) a += s_red[tid * 32 + l];
        emb[(size_t)img * 5 + tid] = a;
    }
}

// ---------------------------------------------------------------------------
// Kernel 2: fused GRU encoder (30 steps) + autoregressive decoder (20 steps)
// One workgroup (4 waves) owns MB=16 batch rows; gate GEMMs via
// v_wmma_f32_16x16x32_f16: N=192 gates = 12 tiles, 3 per wave; K(h)=64 = 2
// k-steps, K(x) = 1 padded k-step.  Weights live in VGPRs across all steps.
// ---------------------------------------------------------------------------
__global__ __launch_bounds__(128) void gru_fused_kernel(
    const float* __restrict__ bbox,     // [B,31,4]
    const float* __restrict__ emb,      // [B*30,5]
    const float* __restrict__ enc_wih, const float* __restrict__ enc_whh,
    const float* __restrict__ enc_bih, const float* __restrict__ enc_bhh,
    const float* __restrict__ dec_wih, const float* __restrict__ dec_whh,
    const float* __restrict__ dec_bih, const float* __restrict__ dec_bhh,
    const float* __restrict__ lin_w,   const float* __restrict__ lin_b,
    float* __restrict__ out)            // [B,20,4]
{
    __shared__ _Float16 sh_h16[MB * GRU_H];
    __shared__ float    sh_h32[MB * GRU_H];
    __shared__ _Float16 sh_x16[MB * 32];        // zero-padded x tile (K=32)
    __shared__ float    sh_gi[MB * NGATE];
    __shared__ float    sh_gh[MB * NGATE];
    __shared__ float    sh_xprev[MB * 4];
    __shared__ float    sh_cum[MB * 4];

    const int tid  = threadIdx.x;
    const int lane = tid & 31;
    const int wave = tid >> 5;
    const int b0   = blockIdx.x * MB;

    for (int i = tid; i < MB * GRU_H; i += 128) { sh_h32[i] = 0.f; sh_h16[i] = (_Float16)0.f; }
    for (int i = tid; i < MB * 32;    i += 128) sh_x16[i] = (_Float16)0.f;
    __syncthreads();

    // ---- encoder weights -> VGPR B-operands (per wave: gate cols 48*wave..) --
    v16h Bh[3][2], Bx[3];
#pragma unroll
    for (int i = 0; i < 3; ++i) {
        int nc = (wave * 3 + i) * 16;
        Bh[i][0] = load_B32x16(enc_whh, GRU_H, nc, 0,  GRU_H, NGATE, lane);
        Bh[i][1] = load_B32x16(enc_whh, GRU_H, nc, 32, GRU_H, NGATE, lane);
        Bx[i]    = load_B32x16(enc_wih, 9,     nc, 0,  9,     NGATE, lane);
    }

    // ---- encoder scan ----
    for (int t = 0; t < ENC_T; ++t) {
        // stage x_t = [diff(4) | emb(5) | 0...] as f16
        for (int i = tid; i < MB * 9; i += 128) {
            int row = i / 9, c = i % 9;
            int bb = b0 + row;
            float v;
            if (c < 4) v = bbox[((size_t)bb * 31 + t + 1) * 4 + c] -
                           bbox[((size_t)bb * 31 + t)     * 4 + c];
            else       v = emb[((size_t)bb * 30 + t) * 5 + (c - 4)];
            sh_x16[row * 32 + c] = (_Float16)v;
        }
        __syncthreads();

        v16h ah0 = load_A16x32(sh_h16,      GRU_H, lane);   // K 0..31
        v16h ah1 = load_A16x32(sh_h16 + 32, GRU_H, lane);   // K 32..63
        v16h ax  = load_A16x32(sh_x16,      32,    lane);
#pragma unroll
        for (int i = 0; i < 3; ++i) {
            int nc = (wave * 3 + i) * 16;
            v8f acch = {};
            acch = __builtin_amdgcn_wmma_f32_16x16x32_f16(false, ah0, false, Bh[i][0],
                                                          (short)0, acch, false, false);
            acch = __builtin_amdgcn_wmma_f32_16x16x32_f16(false, ah1, false, Bh[i][1],
                                                          (short)0, acch, false, false);
            store_acc16x16(sh_gh, NGATE, nc, acch, lane);
            v8f acci = {};
            acci = __builtin_amdgcn_wmma_f32_16x16x32_f16(false, ax, false, Bx[i],
                                                          (short)0, acci, false, false);
            store_acc16x16(sh_gi, NGATE, nc, acci, lane);
        }
        __syncthreads();
        gru_update(sh_gi, sh_gh, sh_h32, sh_h16, enc_bih, enc_bhh, tid);
        __syncthreads();
    }

    // ---- swap in decoder weights ----
#pragma unroll
    for (int i = 0; i < 3; ++i) {
        int nc = (wave * 3 + i) * 16;
        Bh[i][0] = load_B32x16(dec_whh, GRU_H, nc, 0,  GRU_H, NGATE, lane);
        Bh[i][1] = load_B32x16(dec_whh, GRU_H, nc, 32, GRU_H, NGATE, lane);
        Bx[i]    = load_B32x16(dec_wih, 4,     nc, 0,  4,     NGATE, lane);
    }
    // x_prev = diffs[:, -1]; cum = 0; clear emb columns of the x tile
    for (int i = tid; i < MB * 4; i += 128) {
        int row = i >> 2, c = i & 3;
        int bb = b0 + row;
        float v = bbox[((size_t)bb * 31 + 30) * 4 + c] -
                  bbox[((size_t)bb * 31 + 29) * 4 + c];
        sh_xprev[i] = v; sh_cum[i] = 0.f;
        sh_x16[row * 32 + c] = (_Float16)v;
    }
    for (int i = tid; i < MB * 5; i += 128) {
        int row = i / 5, c = 4 + i % 5;
        sh_x16[row * 32 + c] = (_Float16)0.f;
    }
    __syncthreads();

    // ---- autoregressive decoder ----
    for (int s = 0; s < DEC_T; ++s) {
        v16h ah0 = load_A16x32(sh_h16,      GRU_H, lane);
        v16h ah1 = load_A16x32(sh_h16 + 32, GRU_H, lane);
        v16h ax  = load_A16x32(sh_x16,      32,    lane);
#pragma unroll
        for (int i = 0; i < 3; ++i) {
            int nc = (wave * 3 + i) * 16;
            v8f acch = {};
            acch = __builtin_amdgcn_wmma_f32_16x16x32_f16(false, ah0, false, Bh[i][0],
                                                          (short)0, acch, false, false);
            acch = __builtin_amdgcn_wmma_f32_16x16x32_f16(false, ah1, false, Bh[i][1],
                                                          (short)0, acch, false, false);
            store_acc16x16(sh_gh, NGATE, nc, acch, lane);
            v8f acci = {};
            acci = __builtin_amdgcn_wmma_f32_16x16x32_f16(false, ax, false, Bx[i],
                                                          (short)0, acci, false, false);
            store_acc16x16(sh_gi, NGATE, nc, acci, lane);
        }
        __syncthreads();
        gru_update(sh_gi, sh_gh, sh_h32, sh_h16, dec_bih, dec_bhh, tid);
        __syncthreads();

        // x_new = h @ lin_w^T + lin_b + x_prev ; cumsum ; emit cum + offset
        if (tid < MB * 4) {
            int row = tid >> 2, c = tid & 3;
            const float* hr = sh_h32 + row * GRU_H;
            float acc = lin_b[c];
            for (int k = 0; k < GRU_H; ++k) acc += hr[k] * lin_w[c * GRU_H + k];
            float xn = acc + sh_xprev[tid];
            float cm = sh_cum[tid] + xn;
            sh_cum[tid] = cm; sh_xprev[tid] = xn;
            sh_x16[row * 32 + c] = (_Float16)xn;
            int bb = b0 + row;
            out[((size_t)bb * DEC_T + s) * 4 + c] =
                cm + bbox[((size_t)bb * 31 + 30) * 4 + c];
        }
        __syncthreads();
    }
}

// ---------------------------------------------------------------------------
extern "C" void kernel_launch(void* const* d_in, const int* in_sizes, int n_in,
                              void* d_out, int out_size, void* d_ws, size_t ws_size,
                              hipStream_t stream) {
    const float* bbox    = (const float*)d_in[0];
    const float* head    = (const float*)d_in[1];
    const float* conv1_w = (const float*)d_in[2];
    const float* conv1_b = (const float*)d_in[3];
    const float* conv2_w = (const float*)d_in[4];
    const float* conv2_b = (const float*)d_in[5];
    const float* fc_w    = (const float*)d_in[6];
    const float* fc_b    = (const float*)d_in[7];
    const float* enc_wih = (const float*)d_in[8];
    const float* enc_whh = (const float*)d_in[9];
    const float* enc_bih = (const float*)d_in[10];
    const float* enc_bhh = (const float*)d_in[11];
    const float* dec_wih = (const float*)d_in[12];
    const float* dec_whh = (const float*)d_in[13];
    const float* dec_bih = (const float*)d_in[14];
    const float* dec_bhh = (const float*)d_in[15];
    const float* lin_w   = (const float*)d_in[16];
    const float* lin_b   = (const float*)d_in[17];

    const int B = 1024, L = 30;
    float* emb = (float*)d_ws;                     // [B*L, 5]

    conv_head_kernel<<<B * L, 256, 0, stream>>>(head, conv1_w, conv1_b,
                                                conv2_w, conv2_b, fc_w, fc_b, emb);

    gru_fused_kernel<<<B / MB, 128, 0, stream>>>(bbox, emb,
                                                 enc_wih, enc_whh, enc_bih, enc_bhh,
                                                 dec_wih, dec_whh, dec_bih, dec_bhh,
                                                 lin_w, lin_b, (float*)d_out);
}
